// HGRSigma_10754598109736
// MI455X (gfx1250) — compile-verified, matrix operand order checked by reading, not run on hardware
//
#include <hip/hip_runtime.h>

// ---------------------------------------------------------------------------
// HGR hypergraph regression forward for MI455X (gfx1250, wave32, WMMA)
// Dominant op: 16384x16384x512 Gram matrix (275 GFLOP) fused with top-16
// selection, computed with v_wmma_f32_16x16x32_bf16, register-blocked 4
// column tiles per A-fragment load (2.5 b128 loads / wmma).
// ---------------------------------------------------------------------------

#define NNODES 16384
#define IN_CH  512
#define HID    128
#define KNN    16
#define NT     32
#define SLOPE  0.01f
#define GRAM_SPLITS 8           // column-tile splits for the fused gram+topk
#define CBLK   4                // column tiles per wave iteration (reg blocking)
#define CAND   (GRAM_SPLITS * KNN)

typedef __attribute__((ext_vector_type(8)))  __bf16 v8bf;
typedef __attribute__((ext_vector_type(16))) __bf16 v16bf;
typedef __attribute__((ext_vector_type(8)))  float  v8f;

// float -> bf16 with round-to-nearest-even (bit manipulation; no __bf16 math)
static __device__ __forceinline__ __bf16 f2bf(float f) {
  unsigned u = __builtin_bit_cast(unsigned, f);
  u += 0x7fffu + ((u >> 16) & 1u);
  unsigned short s = (unsigned short)(u >> 16);
  return __builtin_bit_cast(__bf16, s);
}

// 16-bit A/B fragment for V_WMMA_*_16x16x32: lane holds row/col (lane&15),
// elements 0..7  <-> K = half*8 + 0..7, elements 8..15 <-> K = 16+half*8+0..7
// -> two contiguous 16-byte loads per k-step.
static __device__ __forceinline__ v16bf load_frag16(const __bf16* base, int half) {
  const v8bf lo = *reinterpret_cast<const v8bf*>(base + half * 8);
  const v8bf hi = *reinterpret_cast<const v8bf*>(base + half * 8 + 16);
  return __builtin_shufflevector(lo, hi, 0,1,2,3,4,5,6,7,8,9,10,11,12,13,14,15);
}

// ---------------------------------------------------------------------------
// Prep: bf16 copy of x + fp32 squared norms. One block (128 thr) per row.
// ---------------------------------------------------------------------------
__global__ void prep_x_kernel(const float* __restrict__ x,
                              __bf16* __restrict__ xh,
                              float* __restrict__ sq) {
  const int row = blockIdx.x;
  const int t = threadIdx.x;
  __shared__ float red[128];
  float s = 0.f;
#pragma unroll
  for (int j = 0; j < IN_CH / 128; ++j) {
    const int c = t + j * 128;
    const float v = x[(size_t)row * IN_CH + c];
    xh[(size_t)row * IN_CH + c] = f2bf(v);
    s += v * v;
  }
  red[t] = s;
  __syncthreads();
  for (int st = 64; st > 0; st >>= 1) {
    if (t < st) red[t] += red[t + st];
    __syncthreads();
  }
  if (t == 0) sq[row] = red[0];
}

// Transpose + convert weights: WT[n*K + k] = bf16(W[k*N + n])
__global__ void wconv_kernel(const float* __restrict__ W, __bf16* __restrict__ WT,
                             int Kdim, int Ncols) {
  const int i = blockIdx.x * blockDim.x + threadIdx.x;
  if (i >= Kdim * Ncols) return;
  const int k = i / Ncols, n = i % Ncols;
  WT[(size_t)n * Kdim + k] = f2bf(W[i]);
}

// ---------------------------------------------------------------------------
// Generic bf16 WMMA GEMM: C[M x Nc] = A[M x Kd] @ BT[Nc x Kd]^T, f32 accum.
// One wave per 16x16 tile.
// ---------------------------------------------------------------------------
__global__ void gemm_bf16_wmma_kernel(const __bf16* __restrict__ A,
                                      const __bf16* __restrict__ BT,
                                      float* __restrict__ C,
                                      int M, int Nc, int Kd) {
  const int ntn = Nc >> 4;
  const int mt = blockIdx.x / ntn;
  const int nt = blockIdx.x % ntn;
  const int lane = threadIdx.x & 31;
  const int l16 = lane & 15, half = lane >> 4;
  const __bf16* arow = A  + (size_t)(mt * 16 + l16) * Kd;
  const __bf16* brow = BT + (size_t)(nt * 16 + l16) * Kd;
  v8f acc = {};
  for (int ks = 0; ks < Kd; ks += 32) {
    const v16bf af = load_frag16(arow + ks, half);
    const v16bf bf = load_frag16(brow + ks, half);
    acc = __builtin_amdgcn_wmma_f32_16x16x32_bf16(false, af, false, bf,
                                                  (short)0, acc, false, false);
  }
  // C/D layout: VGPR r -> (M = r + 8*half, N = l16)
  float* crow = C + (size_t)(mt * 16 + 8 * half) * Nc + nt * 16 + l16;
#pragma unroll
  for (int r = 0; r < 8; ++r) crow[(size_t)r * Nc] = acc[r];
}

// ---------------------------------------------------------------------------
// Fused Gram (bf16 WMMA, full K=512) + per-row top-16 selection.
// grid = (N/16, GRAM_SPLITS), block = 32 (one wave). Each iteration computes
// CBLK 16x16 tiles sharing one A fragment per k-step (register blocking),
// stages the 16 x (16*CBLK) f32 slab through LDS, then lanes 0-15 update
// per-row sorted top-16 lists.
// ---------------------------------------------------------------------------
__global__ void gram_topk_kernel(const __bf16* __restrict__ xh,
                                 const float* __restrict__ sq,
                                 float* __restrict__ cand_d,
                                 int* __restrict__ cand_i) {
  __shared__ float dist[16][CBLK * 16 + 1];
  __shared__ float sq_r[16];
  const int lane = threadIdx.x & 31;
  const int l16 = lane & 15, half = lane >> 4;
  const int rbase = blockIdx.x * 16;
  const int split = blockIdx.y;
  if (lane < 16) sq_r[lane] = sq[rbase + lane];

  float bd[KNN];
  int   bi[KNN];
#pragma unroll
  for (int j = 0; j < KNN; ++j) { bd[j] = 3.0e38f; bi[j] = 0; }

  const __bf16* arow = xh + (size_t)(rbase + l16) * IN_CH;
  const int tiles_per_split = NNODES / 16 / GRAM_SPLITS;       // 128
  const int ct0 = split * tiles_per_split;

  for (int ctg = 0; ctg < tiles_per_split / CBLK; ++ctg) {
    const int cbase = (ct0 + ctg * CBLK) * 16;
    const __bf16* brow[CBLK];
#pragma unroll
    for (int q = 0; q < CBLK; ++q)
      brow[q] = xh + (size_t)(cbase + q * 16 + l16) * IN_CH;

    v8f acc[CBLK];
#pragma unroll
    for (int q = 0; q < CBLK; ++q) acc[q] = {};

    for (int ks = 0; ks < IN_CH; ks += 32) {
      const v16bf af = load_frag16(arow + ks, half);
#pragma unroll
      for (int q = 0; q < CBLK; ++q) {
        const v16bf bf = load_frag16(brow[q] + ks, half);
        acc[q] = __builtin_amdgcn_wmma_f32_16x16x32_bf16(false, af, false, bf,
                                                         (short)0, acc[q], false, false);
      }
    }
    __syncthreads();   // protect previous iteration's selection reads
#pragma unroll
    for (int q = 0; q < CBLK; ++q)
#pragma unroll
      for (int r = 0; r < 8; ++r) dist[8 * half + r][q * 16 + l16] = acc[q][r];
    __syncthreads();
    if (lane < 16) {
      const float sr = sq_r[lane];
#pragma unroll 4
      for (int n = 0; n < CBLK * 16; ++n) {
        const int col = cbase + n;
        const float d = sr + sq[col] - 2.0f * dist[lane][n];
        if (d < bd[KNN - 1]) {           // insert keeping ascending order
          bd[KNN - 1] = d; bi[KNN - 1] = col;
          for (int j = KNN - 1; j > 0 && bd[j] < bd[j - 1]; --j) {
            const float td = bd[j]; bd[j] = bd[j - 1]; bd[j - 1] = td;
            const int   ti = bi[j]; bi[j] = bi[j - 1]; bi[j - 1] = ti;
          }
        }
      }
    }
  }
  if (lane < 16) {
    const size_t base = (size_t)(rbase + lane) * CAND + split * KNN;
#pragma unroll
    for (int j = 0; j < KNN; ++j) { cand_d[base + j] = bd[j]; cand_i[base + j] = bi[j]; }
  }
}

// Merge GRAM_SPLITS partial top-16 lists -> final top-16 per row.
__global__ void knn_merge_kernel(const float* __restrict__ cand_d,
                                 const int* __restrict__ cand_i,
                                 int* __restrict__ nn) {
  const int row = blockIdx.x * blockDim.x + threadIdx.x;
  if (row >= NNODES) return;
  float bd[KNN]; int bi[KNN];
#pragma unroll
  for (int j = 0; j < KNN; ++j) { bd[j] = 3.0e38f; bi[j] = 0; }
  const size_t base = (size_t)row * CAND;
  for (int t = 0; t < CAND; ++t) {
    const float d = cand_d[base + t];
    if (d < bd[KNN - 1]) {
      bd[KNN - 1] = d; bi[KNN - 1] = cand_i[base + t];
      for (int j = KNN - 1; j > 0 && bd[j] < bd[j - 1]; --j) {
        const float td = bd[j]; bd[j] = bd[j - 1]; bd[j - 1] = td;
        const int   ti = bi[j]; bi[j] = bi[j - 1]; bi[j - 1] = ti;
      }
    }
  }
#pragma unroll
  for (int j = 0; j < KNN; ++j) nn[(size_t)row * KNN + j] = bi[j];
}

// Node degree Dv (exact: integer-valued float adds)
__global__ void dv_kernel(const int* __restrict__ nn, float* __restrict__ Dv) {
  const int i = blockIdx.x * blockDim.x + threadIdx.x;
  if (i < NNODES * KNN) atomicAdd(&Dv[nn[i]], 1.0f);
}

// e[i] = mean over K neighbors of h (De == K). block per node, thread per ch.
__global__ void gather_e_kernel(const float* __restrict__ h,
                                const int* __restrict__ nn,
                                float* __restrict__ e) {
  __shared__ int nns[KNN];
  const int i = blockIdx.x, c = threadIdx.x;
  if (c < KNN) nns[c] = nn[(size_t)i * KNN + c];
  __syncthreads();
  float s = 0.f;
#pragma unroll
  for (int j = 0; j < KNN; ++j) s += h[(size_t)nns[j] * HID + c];
  e[(size_t)i * HID + c] = s * (1.0f / (float)KNN);
}

// v[n] += e[i] for each edge i containing n (scatter; normalized later)
__global__ void scatter_v_kernel(const float* __restrict__ e,
                                 const int* __restrict__ nn,
                                 float* __restrict__ v) {
  __shared__ int nns[KNN];
  const int i = blockIdx.x, c = threadIdx.x;
  if (c < KNN) nns[c] = nn[(size_t)i * KNN + c];
  __syncthreads();
  const float ec = e[(size_t)i * HID + c];
#pragma unroll
  for (int j = 0; j < KNN; ++j) atomicAdd(&v[(size_t)nns[j] * HID + c], ec);
}

// out = leaky_relu(v/Dv + bias); write f32 + bf16 (bf16 feeds next WMMA GEMM)
__global__ void finalize_kernel(const float* __restrict__ v,
                                const float* __restrict__ Dv,
                                const float* __restrict__ bias,
                                float* __restrict__ hf,
                                __bf16* __restrict__ hb) {
  const int i = blockIdx.x, c = threadIdx.x;
  float val = v[(size_t)i * HID + c] / Dv[i] + bias[c];
  val = (val >= 0.f) ? val : SLOPE * val;
  hf[(size_t)i * HID + c] = val;
  hb[(size_t)i * HID + c] = f2bf(val);
}

// Deterministic pooling: per-block column partials, then fixed-order reduce.
__global__ void pool_partial_kernel(const float* __restrict__ feats,
                                    float* __restrict__ part) {
  const int b = blockIdx.x, c = threadIdx.x;
  float s = 0.f;
  const float* base = feats + (size_t)b * 128 * HID + c;
  for (int r = 0; r < 128; ++r) s += base[(size_t)r * HID];
  part[(size_t)b * HID + c] = s;
}

__global__ void head_kernel(const float* __restrict__ part,
                            const float* __restrict__ w_mean, const float* __restrict__ b_mean,
                            const float* __restrict__ w_alpha, const float* __restrict__ b_alpha,
                            float* __restrict__ out_mean, float* __restrict__ out_alpha,
                            float* __restrict__ out_pool) {
  __shared__ float fp[HID];
  const int t = threadIdx.x;
  float s = 0.f;
  for (int b = 0; b < NNODES / 128; ++b) s += part[(size_t)b * HID + t];
  const float p = s * (1.0f / (float)NNODES);
  fp[t] = p;
  out_pool[t] = p;
  __syncthreads();
  if (t < NT) {
    float m = b_mean[t], a = b_alpha[t];
    for (int c = 0; c < HID; ++c) {
      m += fp[c] * w_mean[(size_t)c * NT + t];
      a += fp[c] * w_alpha[(size_t)c * NT + t];
    }
    out_mean[t] = 1.0f / (1.0f + expf(-m));
    out_alpha[t] = (a < 0.f) ? (a - log1pf(expf(a))) : (-log1pf(expf(-a)));
  }
}

// ---------------------------------------------------------------------------
extern "C" void kernel_launch(void* const* d_in, const int* in_sizes, int n_in,
                              void* d_out, int out_size, void* d_ws, size_t ws_size,
                              hipStream_t stream) {
  const float* x       = (const float*)d_in[0];
  const float* theta0  = (const float*)d_in[1];
  const float* bias0   = (const float*)d_in[2];
  const float* theta1  = (const float*)d_in[3];
  const float* bias1   = (const float*)d_in[4];
  const float* w_mean  = (const float*)d_in[5];
  const float* b_mean  = (const float*)d_in[6];
  const float* w_alpha = (const float*)d_in[7];
  const float* b_alpha = (const float*)d_in[8];

  float* out       = (float*)d_out;
  float* out_mean  = out;
  float* out_alpha = out + NT;
  float* feats     = out + 2 * NT;
  float* out_pool  = out + 2 * NT + (size_t)NNODES * HID;

  // workspace carve-up (256B aligned)
  char* ws = (char*)d_ws;
  size_t o = 0;
  auto carve = [&](size_t bytes) -> char* {
    char* p = ws + o;
    o = (o + bytes + 255) & ~(size_t)255;
    return p;
  };
  __bf16* xh     = (__bf16*)carve((size_t)NNODES * IN_CH * 2);
  float*  sq     = (float*) carve((size_t)NNODES * 4);
  int*    nn     = (int*)   carve((size_t)NNODES * KNN * 4);
  float*  cand_d = (float*) carve((size_t)NNODES * CAND * 4);
  int*    cand_i = (int*)   carve((size_t)NNODES * CAND * 4);
  float*  Dv     = (float*) carve((size_t)NNODES * 4);
  __bf16* th0T   = (__bf16*)carve((size_t)IN_CH * HID * 2);
  __bf16* th1T   = (__bf16*)carve((size_t)HID * HID * 2);
  float*  g      = (float*) carve((size_t)NNODES * HID * 4);
  float*  e      = (float*) carve((size_t)NNODES * HID * 4);
  float*  v      = (float*) carve((size_t)NNODES * HID * 4);
  float*  h1f    = (float*) carve((size_t)NNODES * HID * 4);
  __bf16* hbf    = (__bf16*)carve((size_t)NNODES * HID * 2);
  float*  part   = (float*) carve((size_t)(NNODES / 128) * HID * 4);

  // 1) prep
  prep_x_kernel<<<NNODES, 128, 0, stream>>>(x, xh, sq);
  wconv_kernel<<<(IN_CH * HID + 255) / 256, 256, 0, stream>>>(theta0, th0T, IN_CH, HID);
  wconv_kernel<<<(HID * HID + 255) / 256, 256, 0, stream>>>(theta1, th1T, HID, HID);

  // 2) fused Gram (bf16 WMMA, 4-wide column blocking) + top-16
  gram_topk_kernel<<<dim3(NNODES / 16, GRAM_SPLITS), 32, 0, stream>>>(xh, sq, cand_d, cand_i);
  knn_merge_kernel<<<(NNODES + 255) / 256, 256, 0, stream>>>(cand_d, cand_i, nn);
  hipMemsetAsync(Dv, 0, (size_t)NNODES * 4, stream);
  dv_kernel<<<(NNODES * KNN + 255) / 256, 256, 0, stream>>>(nn, Dv);

  // 3) hyconv layer 1
  gemm_bf16_wmma_kernel<<<(NNODES / 16) * (HID / 16), 32, 0, stream>>>(xh, th0T, g, NNODES, HID, IN_CH);
  gather_e_kernel<<<NNODES, HID, 0, stream>>>(g, nn, e);
  hipMemsetAsync(v, 0, (size_t)NNODES * HID * 4, stream);
  scatter_v_kernel<<<NNODES, HID, 0, stream>>>(e, nn, v);
  finalize_kernel<<<NNODES, HID, 0, stream>>>(v, Dv, bias0, h1f, hbf);

  // 4) hyconv layer 2 (feats -> d_out)
  gemm_bf16_wmma_kernel<<<(NNODES / 16) * (HID / 16), 32, 0, stream>>>(hbf, th1T, g, NNODES, HID, HID);
  gather_e_kernel<<<NNODES, HID, 0, stream>>>(g, nn, e);
  hipMemsetAsync(v, 0, (size_t)NNODES * HID * 4, stream);
  scatter_v_kernel<<<NNODES, HID, 0, stream>>>(e, nn, v);
  finalize_kernel<<<NNODES, HID, 0, stream>>>(v, Dv, bias1, feats, hbf);

  // 5) deterministic pooling + heads
  pool_partial_kernel<<<NNODES / 128, HID, 0, stream>>>(feats, part);
  head_kernel<<<1, HID, 0, stream>>>(part, w_mean, b_mean, w_alpha, b_alpha,
                                     out_mean, out_alpha, out_pool);
}
